// MultiHeadAttention_72241349919166
// MI455X (gfx1250) — matrix operand order repaired and checked by reading, not compiled, not run on hardware
//
#include <hip/hip_runtime.h>

typedef _Float16 h16;
typedef __attribute__((ext_vector_type(16))) _Float16 v16h;
typedef __attribute__((ext_vector_type(8)))  _Float16 v8h;
typedef __attribute__((ext_vector_type(4)))  _Float16 v4h;
typedef __attribute__((ext_vector_type(8)))  float    v8f;
typedef __attribute__((ext_vector_type(4)))  float    v4f;

#define SEQ   2048
#define DM    1024
#define NH    16
#define DK    64
#define BATCH 2
#define LDP   520   // 512 + 8 pad halves -> row pitch 1040B = 65*16B (bank-spread)

// ------------------------------------------------------------------
// fp32 -> fp16 conversion (4 elems/thread, all sizes are multiples of 1024)
// ------------------------------------------------------------------
__global__ void __launch_bounds__(256)
cvt_f32_f16(const float* __restrict__ s, h16* __restrict__ d, int n4) {
  int i = blockIdx.x * 256 + threadIdx.x;
  if (i >= n4) return;
  v4f x = ((const v4f*)s)[i];
  v4h y;
  y.x = (h16)x.x; y.y = (h16)x.y; y.z = (h16)x.z; y.w = (h16)x.w;
  ((v4h*)d)[i] = y;
}

// ------------------------------------------------------------------
// WMMA operand loads, contiguous k order (k = 16*(lane>>4) + s), shared by A and B.
// A (16x32 f16): lane<16 holds K{0..15}, lane>=16 K{16..31};
// caller passes p = row_ptr + k + (lane>>4)*16 ; second chunk is p+8.
// B (32x16 f16): lane holds column N=lane&15, K run (lane>>4)*16..+15 ;
// caller passes p = col_row_ptr + k + (lane>>4)*16 (16 contiguous halves).
// Works for both global and LDS pointers (compiler picks global_/ds_ loads).
// ------------------------------------------------------------------
__device__ __forceinline__ v16h ldA(const h16* __restrict__ p) {
  v8h lo = *(const v8h*)p;
  v8h hi = *(const v8h*)(p + 8);
  return __builtin_shufflevector(lo, hi, 0,1,2,3,4,5,6,7,8,9,10,11,12,13,14,15);
}
__device__ __forceinline__ v16h ldB(const h16* __restrict__ p) {
  v8h lo = *(const v8h*)p;
  v8h hi = *(const v8h*)(p + 8);
  return __builtin_shufflevector(lo, hi, 0,1,2,3,4,5,6,7,8,9,10,11,12,13,14,15);
}

// ------------------------------------------------------------------
// 64x64 output tile per wave: 4x4 grid of 16x16 WMMA accumulators.
// ------------------------------------------------------------------
__device__ __forceinline__ void mma64x64(const h16* __restrict__ A, int lda,
                                         const h16* __restrict__ B, int ldb,
                                         int K, v8f acc[4][4]) {
  const int lane = threadIdx.x & 31;
  const int l = lane & 15, hi = lane >> 4;
  for (int k = 0; k < K; k += 32) {
    if (k + 256 < K) {            // uniform branch; emits global_prefetch_b8
      __builtin_prefetch(A + (size_t)l * lda + k + 256, 0, 1);
      __builtin_prefetch(B + (size_t)l * ldb + k + 256, 0, 1);
    }
    v16h af[4], bf[4];
#pragma unroll
    for (int i = 0; i < 4; ++i)
      af[i] = ldA(A + (size_t)(i * 16 + l) * lda + k + hi * 16);
#pragma unroll
    for (int j = 0; j < 4; ++j)
      bf[j] = ldB(B + (size_t)(j * 16 + l) * ldb + k + hi * 16);
#pragma unroll
    for (int i = 0; i < 4; ++i)
#pragma unroll
      for (int j = 0; j < 4; ++j)
        acc[i][j] = __builtin_amdgcn_wmma_f32_16x16x32_f16(
            false, af[i], false, bf[j], (short)0, acc[i][j], false, false);
  }
}

// ------------------------------------------------------------------
// Projection GEMM, f16 output row-major:  out = x @ W^T + b
// ------------------------------------------------------------------
__global__ void __launch_bounds__(256)
gemm_bias_f16out(const h16* __restrict__ A, const h16* __restrict__ W,
                 const float* __restrict__ bias, h16* __restrict__ out,
                 int M, int N, int K) {
  int tile = blockIdx.x * 8 + (threadIdx.x >> 5);
  int tN = N >> 6;
  int tm = tile / tN, tn = tile - tm * tN;
  int m0 = tm << 6, n0 = tn << 6;
  v8f acc[4][4] = {};
  mma64x64(A + (size_t)m0 * K, K, W + (size_t)n0 * K, K, K, acc);
  const int lane = threadIdx.x & 31;
  const int l = lane & 15, hi = lane >> 4;
#pragma unroll
  for (int i = 0; i < 4; ++i)
#pragma unroll
    for (int j = 0; j < 4; ++j) {
      int col = n0 + j * 16 + l;
      float bv = bias[col];
#pragma unroll
      for (int r = 0; r < 8; ++r) {
        int row = m0 + i * 16 + 8 * hi + r;   // C layout: VGPR r -> M=r (+8 for hi lanes)
        out[(size_t)row * N + col] = (h16)(acc[i][j][r] + bv);
      }
    }
}

// Final GEMM, fp32 output:  out = ctx @ Wo^T + bo
__global__ void __launch_bounds__(256)
gemm_bias_f32out(const h16* __restrict__ A, const h16* __restrict__ W,
                 const float* __restrict__ bias, float* __restrict__ out,
                 int M, int N, int K) {
  int tile = blockIdx.x * 8 + (threadIdx.x >> 5);
  int tN = N >> 6;
  int tm = tile / tN, tn = tile - tm * tN;
  int m0 = tm << 6, n0 = tn << 6;
  v8f acc[4][4] = {};
  mma64x64(A + (size_t)m0 * K, K, W + (size_t)n0 * K, K, K, acc);
  const int lane = threadIdx.x & 31;
  const int l = lane & 15, hi = lane >> 4;
#pragma unroll
  for (int i = 0; i < 4; ++i)
#pragma unroll
    for (int j = 0; j < 4; ++j) {
      int col = n0 + j * 16 + l;
      float bv = bias[col];
#pragma unroll
      for (int r = 0; r < 8; ++r) {
        int row = m0 + i * 16 + 8 * hi + r;
        out[(size_t)row * N + col] = acc[i][j][r] + bv;
      }
    }
}

// ------------------------------------------------------------------
// V projection with per-head transposed store: VT[b][h][d][s]
// ------------------------------------------------------------------
__global__ void __launch_bounds__(256)
gemm_bias_vT(const h16* __restrict__ A, const h16* __restrict__ W,
             const float* __restrict__ bias, h16* __restrict__ VT) {
  int tile = blockIdx.x * 8 + (threadIdx.x >> 5);
  const int tN = DM >> 6;               // 16
  int tm = tile / tN, tn = tile - tm * tN;
  int m0 = tm << 6, n0 = tn << 6;
  v8f acc[4][4] = {};
  mma64x64(A + (size_t)m0 * DM, DM, W + (size_t)n0 * DM, DM, DM, acc);
  const int lane = threadIdx.x & 31;
  const int l = lane & 15, hi = lane >> 4;
  int b = m0 >> 11;                     // tiles never straddle the batch boundary
  int jb = m0 & (SEQ - 1);
#pragma unroll
  for (int i = 0; i < 4; ++i)
#pragma unroll
    for (int j = 0; j < 4; ++j) {
      int col = n0 + j * 16 + l;        // = h*64 + d
      float bv = bias[col];
      v8h pk;
#pragma unroll
      for (int r = 0; r < 8; ++r) pk[r] = (h16)(acc[i][j][r] + bv);
      size_t off = ((size_t)(b * NH * DK + col)) * SEQ + jb + i * 16 + 8 * hi;
      *(v8h*)(VT + off) = pk;
    }
}

// ------------------------------------------------------------------
// Fully fused attention: QK^T + masked softmax + (attention @ V).
// One block owns a 32-row x 2048-col strip of one (b,h) score matrix.
// Wave w covers column tiles {c*8+w : c=0..3}; per chunk c the block
// produces the 32x512 probability slab: pass 1 builds online row (m,s)
// stats in registers, pass 2 recomputes the cheap K=64 score WMMAs,
// streams normalized probabilities to d_out (non-temporal, written
// exactly once, never re-read from HBM), stages them in LDS, and each
// wave immediately multiplies its K=64 slice against V^T, reducing the
// 32x64 context tile with ds_add_f32. Eliminates the 537 MB attention
// re-read of the unfused pipeline.
// ------------------------------------------------------------------
__global__ void __launch_bounds__(256)
attn_fused(const h16* __restrict__ Q, const h16* __restrict__ Kp,
           const h16* __restrict__ VT, const int* __restrict__ mask,
           float* __restrict__ att, h16* __restrict__ ctx) {
  const int bh = blockIdx.x >> 6;           // 64 (b,h) pairs
  const int tm = blockIdx.x & 63;           // 64 strips of 32 rows
  const int b = bh >> 4, h = bh & 15;
  const int w = threadIdx.x >> 5;
  const int lane = threadIdx.x & 31;
  const int l = lane & 15, hi = lane >> 4;

  const h16* A     = Q  + ((size_t)(b * SEQ + tm * 32)) * DM + h * DK;
  const h16* Bbase = Kp + ((size_t)b * SEQ) * DM + h * DK;
  const h16* VTh   = VT + (size_t)bh * DK * SEQ;
  const int* mbase = mask + ((size_t)b * SEQ + tm * 32) * SEQ;
  float*     obase = att + ((size_t)bh * SEQ + tm * 32) * SEQ;

  __shared__ h16   pr[32 * LDP];            // 32x512 prob chunk (padded)
  __shared__ float ctxbuf[32 * 64];         // fp32 context accumulator
  __shared__ float lm[8][32], ls[8][32], gm[32], gs[32];

  // A fragments (K=64 -> 2 k-steps), loaded once, reused by both passes
  v16h af[2][2];
#pragma unroll
  for (int i = 0; i < 2; ++i)
#pragma unroll
    for (int kk = 0; kk < 2; ++kk)
      af[i][kk] = ldA(A + (size_t)(i * 16 + l) * DM + kk * 32 + hi * 16);

  // per-lane running row stats over this lane's own columns
  float rm[2][8], rs[2][8];
#pragma unroll
  for (int i = 0; i < 2; ++i)
#pragma unroll
    for (int r = 0; r < 8; ++r) { rm[i][r] = -3.0e38f; rs[i][r] = 0.f; }

  // ---------------- pass 1: scores -> online (m,s) ----------------
  for (int c = 0; c < 4; ++c) {
    const int col0 = (c * 8 + w) * 64;
    v8f acc[2][4] = {};
    v16h bf[4][2];
#pragma unroll
    for (int j = 0; j < 4; ++j)
#pragma unroll
      for (int kk = 0; kk < 2; ++kk)
        bf[j][kk] = ldB(Bbase + (size_t)(col0 + j * 16 + l) * DM + kk * 32 + hi * 16);
#pragma unroll
    for (int kk = 0; kk < 2; ++kk)
#pragma unroll
      for (int i = 0; i < 2; ++i)
#pragma unroll
        for (int j = 0; j < 4; ++j)
          acc[i][j] = __builtin_amdgcn_wmma_f32_16x16x32_f16(
              false, af[i][kk], false, bf[j][kk], (short)0, acc[i][j], false, false);
#pragma unroll
    for (int i = 0; i < 2; ++i)
#pragma unroll
      for (int r = 0; r < 8; ++r) {
        const int* mrow = mbase + (size_t)(i * 16 + 8 * hi + r) * SEQ + col0;
        float vls[4];
#pragma unroll
        for (int j = 0; j < 4; ++j) {
          float s = acc[i][j][r] * 0.125f;
          if (mrow[j * 16 + l] == 0) s = 1e-9f;   // faithful to reference
          vls[j] = s;
        }
        float mloc = fmaxf(fmaxf(vls[0], vls[1]), fmaxf(vls[2], vls[3]));
        float mnew = fmaxf(rm[i][r], mloc);
        rs[i][r] = rs[i][r] * __expf(rm[i][r] - mnew)
                 + __expf(vls[0] - mnew) + __expf(vls[1] - mnew)
                 + __expf(vls[2] - mnew) + __expf(vls[3] - mnew);
        rm[i][r] = mnew;
      }
  }

  // merge (m,s) across the 16 lanes of each half (rows differ per half,
  // xor masks 1/2/4/8 never cross the 16-lane boundary)
#pragma unroll
  for (int i = 0; i < 2; ++i)
#pragma unroll
    for (int r = 0; r < 8; ++r) {
#pragma unroll
      for (int o = 1; o < 16; o <<= 1) {
        float om = __shfl_xor(rm[i][r], o, 32);
        float os = __shfl_xor(rs[i][r], o, 32);
        float mn = fmaxf(rm[i][r], om);
        rs[i][r] = rs[i][r] * __expf(rm[i][r] - mn) + os * __expf(om - mn);
        rm[i][r] = mn;
      }
    }

  if (l == 0) {
#pragma unroll
    for (int i = 0; i < 2; ++i)
#pragma unroll
      for (int r = 0; r < 8; ++r) {
        int rr = i * 16 + 8 * hi + r;
        lm[w][rr] = rm[i][r];
        ls[w][rr] = rs[i][r];
      }
  }
  // zero the context accumulator while stats land in LDS
#pragma unroll
  for (int e = 0; e < 8; ++e) ctxbuf[threadIdx.x * 8 + e] = 0.f;
  __syncthreads();
  if (threadIdx.x < 32) {
    int rr = threadIdx.x;
    float M = -3.0e38f, S = 0.f;
#pragma unroll
    for (int w2 = 0; w2 < 8; ++w2) {
      float om = lm[w2][rr], os = ls[w2][rr];
      float mn = fmaxf(M, om);
      S = S * __expf(M - mn) + os * __expf(om - mn);
      M = mn;
    }
    gm[rr] = M;
    gs[rr] = S;
  }
  __syncthreads();

  float Mv[2][8], iS[2][8];
#pragma unroll
  for (int i = 0; i < 2; ++i)
#pragma unroll
    for (int r = 0; r < 8; ++r) {
      int rr = i * 16 + 8 * hi + r;
      Mv[i][r] = gm[rr];
      iS[i][r] = 1.0f / gs[rr];
    }

  // ------- pass 2: recompute scores, emit probs, fold into A@V -------
  for (int c = 0; c < 4; ++c) {
    const int col0 = (c * 8 + w) * 64;    // global column of this wave's tile
    const int cl0  = w * 64;              // column inside the LDS chunk
    v8f acc[2][4] = {};
    v16h bf[4][2];
#pragma unroll
    for (int j = 0; j < 4; ++j)
#pragma unroll
      for (int kk = 0; kk < 2; ++kk)
        bf[j][kk] = ldB(Bbase + (size_t)(col0 + j * 16 + l) * DM + kk * 32 + hi * 16);
#pragma unroll
    for (int kk = 0; kk < 2; ++kk)
#pragma unroll
      for (int i = 0; i < 2; ++i)
#pragma unroll
        for (int j = 0; j < 4; ++j)
          acc[i][j] = __builtin_amdgcn_wmma_f32_16x16x32_f16(
              false, af[i][kk], false, bf[j][kk], (short)0, acc[i][j], false, false);
#pragma unroll
    for (int i = 0; i < 2; ++i)
#pragma unroll
      for (int r = 0; r < 8; ++r) {
        const int rr = i * 16 + 8 * hi + r;
        const int* mrow = mbase + (size_t)rr * SEQ + col0;
        float* orow = obase + (size_t)rr * SEQ + col0;
#pragma unroll
        for (int j = 0; j < 4; ++j) {
          float s = acc[i][j][r] * 0.125f;
          if (mrow[j * 16 + l] == 0) s = 1e-9f;
          float p = __expf(s - Mv[i][r]) * iS[i][r];
          __builtin_nontemporal_store(p, orow + j * 16 + l);
          pr[rr * LDP + cl0 + j * 16 + l] = (h16)p;   // LDS stage for A@V
        }
      }
    __syncthreads();   // probs chunk complete in LDS

    // wave w consumes K slice [c*512 + w*64, +64) of attention @ V
    const int jbase = c * 512 + w * 64;
    v16h a2[2][2], b2[4][2];
#pragma unroll
    for (int i = 0; i < 2; ++i)
#pragma unroll
      for (int kk = 0; kk < 2; ++kk)
        a2[i][kk] = ldA(pr + (size_t)(i * 16 + l) * LDP + w * 64 + kk * 32 + hi * 16);
#pragma unroll
    for (int d = 0; d < 4; ++d)
#pragma unroll
      for (int kk = 0; kk < 2; ++kk)
        b2[d][kk] = ldB(VTh + (size_t)(d * 16 + l) * SEQ + jbase + kk * 32 + hi * 16);
    v8f c2[2][4] = {};
#pragma unroll
    for (int kk = 0; kk < 2; ++kk)
#pragma unroll
      for (int i = 0; i < 2; ++i)
#pragma unroll
        for (int d = 0; d < 4; ++d)
          c2[i][d] = __builtin_amdgcn_wmma_f32_16x16x32_f16(
              false, a2[i][kk], false, b2[d][kk], (short)0, c2[i][d], false, false);
#pragma unroll
    for (int i = 0; i < 2; ++i)
#pragma unroll
      for (int d = 0; d < 4; ++d)
#pragma unroll
        for (int r = 0; r < 8; ++r)
          atomicAdd(&ctxbuf[(i * 16 + 8 * hi + r) * 64 + d * 16 + l],
                    c2[i][d][r]);                      // ds_add_f32
    __syncthreads();   // adds done before pr is overwritten
  }

  // write the 32x64 context tile (8 contiguous values per thread)
  {
    int idx = threadIdx.x * 8;
    int row = idx >> 6, col = idx & 63;
    v8h pk;
#pragma unroll
    for (int e = 0; e < 8; ++e) pk[e] = (h16)ctxbuf[idx + e];
    size_t off = ((size_t)(b * SEQ + tm * 32 + row)) * DM + h * DK + col;
    *(v8h*)(ctx + off) = pk;
  }
}

// ------------------------------------------------------------------
extern "C" void kernel_launch(void* const* d_in, const int* in_sizes, int n_in,
                              void* d_out, int out_size, void* d_ws, size_t ws_size,
                              hipStream_t stream) {
  (void)in_sizes; (void)n_in; (void)out_size; (void)ws_size;
  const float* q    = (const float*)d_in[0];
  const float* k    = (const float*)d_in[1];
  const float* v    = (const float*)d_in[2];
  const int*   mask = (const int*)d_in[3];
  const float* Wq   = (const float*)d_in[4];
  const float* bq   = (const float*)d_in[5];
  const float* Wk   = (const float*)d_in[6];
  const float* bk   = (const float*)d_in[7];
  const float* Wv   = (const float*)d_in[8];
  const float* bv   = (const float*)d_in[9];
  const float* Wo   = (const float*)d_in[10];
  const float* bo   = (const float*)d_in[11];

  h16* ws = (h16*)d_ws;
  const size_t XN = (size_t)BATCH * SEQ * DM;   // 4194304
  const size_t WN = (size_t)DM * DM;            // 1048576
  h16* q16  = ws;
  h16* k16  = q16 + XN;
  h16* v16  = k16 + XN;
  h16* wq16 = v16 + XN;
  h16* wk16 = wq16 + WN;
  h16* wv16 = wk16 + WN;
  h16* wo16 = wv16 + WN;
  h16* Qp   = wo16 + WN;
  h16* Kp   = Qp + XN;
  h16* VT   = Kp + XN;
  h16* ctx  = VT + XN;                          // total ~64 MB of d_ws

  float* out = (float*)d_out;
  float* att = out + XN;                        // attention slot: 2*16*2048*2048 f32

  // fp32 -> fp16 staging
  cvt_f32_f16<<<(int)(XN / 4 / 256), 256, 0, stream>>>(q, q16, (int)(XN / 4));
  cvt_f32_f16<<<(int)(XN / 4 / 256), 256, 0, stream>>>(k, k16, (int)(XN / 4));
  cvt_f32_f16<<<(int)(XN / 4 / 256), 256, 0, stream>>>(v, v16, (int)(XN / 4));
  cvt_f32_f16<<<(int)(WN / 4 / 256), 256, 0, stream>>>(Wq, wq16, (int)(WN / 4));
  cvt_f32_f16<<<(int)(WN / 4 / 256), 256, 0, stream>>>(Wk, wk16, (int)(WN / 4));
  cvt_f32_f16<<<(int)(WN / 4 / 256), 256, 0, stream>>>(Wv, wv16, (int)(WN / 4));
  cvt_f32_f16<<<(int)(WN / 4 / 256), 256, 0, stream>>>(Wo, wo16, (int)(WN / 4));

  // projections: (4096/64)*(1024/64)=1024 tiles, 8 waves/block -> 128 blocks
  gemm_bias_f16out<<<128, 256, 0, stream>>>(q16, wq16, bq, Qp, BATCH * SEQ, DM, DM);
  gemm_bias_f16out<<<128, 256, 0, stream>>>(k16, wk16, bk, Kp, BATCH * SEQ, DM, DM);
  gemm_bias_vT   <<<128, 256, 0, stream>>>(v16, wv16, bv, VT);

  // fully fused attention: 64 (b,h) x 64 strips of 32 rows
  attn_fused<<<4096, 256, 0, stream>>>(Qp, Kp, VT, mask, att, ctx);

  // output projection
  gemm_bias_f32out<<<128, 256, 0, stream>>>(ctx, wo16, bo, out, BATCH * SEQ, DM, DM);
}
